// ETGT_FRD_10187662426181
// MI455X (gfx1250) — matrix operand order repaired and checked by reading, not compiled, not run on hardware
//
#include <hip/hip_runtime.h>

#define N_NODES 50000
#define E_EDGES 800000
#define FDIM_IN 197
#define HDIM 256
#define NHEADS 8
#define HEAD_D 32
#define NLAYERS 5
#define FFDIM 1024
#define INV_SCALE 0.17677669529663687f  // 1/sqrt(32)

#define BM 256  // rows per block (8 waves x 32 rows)
#define BN 64   // cols per block

typedef __bf16 bf16_t;
typedef bf16_t v16bf __attribute__((ext_vector_type(16)));
typedef float  v8f   __attribute__((ext_vector_type(8)));

union FragU { uint4 u4[2]; v16bf bf; };

__device__ __forceinline__ unsigned int f32_to_bf16_rne(float f) {
  unsigned int x = __float_as_uint(f);
  x += 0x7fffu + ((x >> 16) & 1u);
  return x >> 16;
}
__device__ __forceinline__ float gelu_exact(float x) {
  return 0.5f * x * (1.0f + erff(x * 0.7071067811865476f));
}
__device__ __forceinline__ int fenc(float f) {
  int i = __float_as_int(f);
  return i >= 0 ? i : (i ^ 0x7fffffff);
}
__device__ __forceinline__ float fdec(int i) {
  return __int_as_float(i >= 0 ? i : (i ^ 0x7fffffff));
}

// ---------------------------------------------------------------------------
// WMMA GEMM: out[M,Nout] = act(A[M,K] @ W[K,Nout] + bias) + resid
// 256 threads = 8 waves; tile 256(M) x 64(N); each wave: 32 rows x 64 cols
// = 2 A-frags x 4 B-frags = 8 x v_wmma_f32_16x16x32_bf16 per K-chunk.
// ---------------------------------------------------------------------------
__global__ __launch_bounds__(256) void k_gemm_wmma(
    const float* __restrict__ A, const float* __restrict__ W,
    const float* __restrict__ bias, const float* __restrict__ resid,
    float* __restrict__ out, int M, int K, int Nout, int act) {
  __shared__ __align__(16) unsigned short sA[BM * 40];  // [row][k], stride 40 halves
  __shared__ __align__(16) unsigned short sB[BN * 40];  // [n][k] transposed
  const int tid  = threadIdx.x;
  const int wave = tid >> 5;
  const int lane = tid & 31;
  const int m0 = blockIdx.x * BM;
  const int n0 = blockIdx.y * BN;

  v8f acc[2][4] = {};

  const int r  = lane & 15;
  const int hs = lane >> 4;
  const int aBase0 = (wave * 32 + r) * 40 + hs * 8;       // A frag rows 0..15
  const int aBase1 = (wave * 32 + 16 + r) * 40 + hs * 8;  // A frag rows 16..31
  const int bKo    = hs * 8;   // B: lanes0-15 K0..7,16..23, lanes16-31 K8..15,24..31

  // Vector-path thread mapping (uniform across loop)
  const int vaKq  = tid & 7;        // A: float4 index within row (k = 4*vaKq)
  const int vbNq  = (tid & 15) * 4; // B: n quad base
  const int vbKp  = tid >> 4;       // B: k pair (k = 2*vbKp, 2*vbKp+1)
  unsigned int* sB32 = (unsigned int*)sB;

  const bool k4 = (K & 3) == 0;
  const int nk = (K + 31) >> 5;
  for (int kb = 0; kb < nk; ++kb) {
    const int k0 = kb << 5;
    __syncthreads();
    if (k4 && (k0 + 32 <= K)) {
      // ---- vectorized, branch-free staging ----
#pragma unroll
      for (int i = 0; i < 8; ++i) {
        const int row = (tid >> 3) + i * 32;          // 0..255
        const int gr  = min(m0 + row, M - 1);         // clamp (rows indep.)
        const float4 f = *(const float4*)(A + (long)gr * K + k0 + vaKq * 4);
        uint2 p;
        p.x = f32_to_bf16_rne(f.x) | (f32_to_bf16_rne(f.y) << 16);
        p.y = f32_to_bf16_rne(f.z) | (f32_to_bf16_rne(f.w) << 16);
        *(uint2*)&sA[row * 40 + vaKq * 4] = p;        // ds_store_b64
      }
      {
        const float* wp = W + (long)(k0 + 2 * vbKp) * Nout + (n0 + vbNq);
        const float4 r0 = *(const float4*)wp;
        const float4 r1 = *(const float4*)(wp + Nout);
        sB32[(vbNq + 0) * 20 + vbKp] = f32_to_bf16_rne(r0.x) | (f32_to_bf16_rne(r1.x) << 16);
        sB32[(vbNq + 1) * 20 + vbKp] = f32_to_bf16_rne(r0.y) | (f32_to_bf16_rne(r1.y) << 16);
        sB32[(vbNq + 2) * 20 + vbKp] = f32_to_bf16_rne(r0.z) | (f32_to_bf16_rne(r1.z) << 16);
        sB32[(vbNq + 3) * 20 + vbKp] = f32_to_bf16_rne(r0.w) | (f32_to_bf16_rne(r1.w) << 16);
      }
    } else {
      // ---- scalar tail (K=197 only): branch-free via clamp + select ----
#pragma unroll
      for (int i = 0; i < 32; ++i) {
        const int idx = tid + i * 256;
        const int row = idx >> 5, kk = idx & 31;
        const int gr = min(m0 + row, M - 1);
        const int gk = k0 + kk;
        float v = A[(long)gr * K + min(gk, K - 1)];
        v = (gk < K) ? v : 0.0f;
        sA[row * 40 + kk] = (unsigned short)f32_to_bf16_rne(v);
      }
#pragma unroll
      for (int i = 0; i < 8; ++i) {
        const int idx = tid + i * 256;
        const int kk = idx >> 6, nn = idx & 63;
        const int gk = k0 + kk;
        float v = W[(long)min(gk, K - 1) * Nout + (n0 + nn)];
        v = (gk < K) ? v : 0.0f;
        sB[nn * 40 + kk] = (unsigned short)f32_to_bf16_rne(v);
      }
    }
    __syncthreads();

    FragU a0, a1;
    a0.u4[0] = *(const uint4*)&sA[aBase0];
    a0.u4[1] = *(const uint4*)&sA[aBase0 + 16];
    a1.u4[0] = *(const uint4*)&sA[aBase1];
    a1.u4[1] = *(const uint4*)&sA[aBase1 + 16];
#pragma unroll
    for (int t = 0; t < 4; ++t) {
      FragU bfu;
      const int bb = (t * 16 + r) * 40 + bKo;
      bfu.u4[0] = *(const uint4*)&sB[bb];
      bfu.u4[1] = *(const uint4*)&sB[bb + 16];
      acc[0][t] = __builtin_amdgcn_wmma_f32_16x16x32_bf16(
          false, a0.bf, false, bfu.bf, (short)0, acc[0][t], false, false);
      acc[1][t] = __builtin_amdgcn_wmma_f32_16x16x32_bf16(
          false, a1.bf, false, bfu.bf, (short)0, acc[1][t], false, false);
    }
  }

  // Epilogue: C/D layout -> VGPR rr: lanes0-15 row rr, lanes16-31 row rr+8
#pragma unroll
  for (int s = 0; s < 2; ++s) {
#pragma unroll
    for (int t = 0; t < 4; ++t) {
      const int gcol = n0 + t * 16 + r;
      const float bv = bias ? bias[gcol] : 0.0f;
#pragma unroll
      for (int rr = 0; rr < 8; ++rr) {
        const int grow = m0 + wave * 32 + s * 16 + hs * 8 + rr;
        if (grow < M) {
          float v = acc[s][t][rr] + bv;
          if (act) v = gelu_exact(v);
          const long o = (long)grow * Nout + gcol;
          if (resid) v += resid[o];
          out[o] = v;
        }
      }
    }
  }
}

// ---------------------------------------------------------------------------
// LayerNorm over H=256: one wave per row; optional residual add / GELU
// ---------------------------------------------------------------------------
__global__ __launch_bounds__(256) void k_ln(
    const float* __restrict__ in, const float* __restrict__ resid,
    const float* __restrict__ g, const float* __restrict__ b,
    float* __restrict__ out, int M, int doGelu) {
  const int row  = blockIdx.x * 8 + (threadIdx.x >> 5);
  const int lane = threadIdx.x & 31;
  if (row >= M) return;
  const long base = (long)row * HDIM;
  float x[8], s = 0.0f, sq = 0.0f;
#pragma unroll
  for (int i = 0; i < 8; ++i) {
    int j = lane + i * 32;
    float v = in[base + j];
    if (resid) v += resid[base + j];
    x[i] = v; s += v; sq += v * v;
  }
#pragma unroll
  for (int o = 16; o > 0; o >>= 1) {
    s  += __shfl_xor(s, o, 32);
    sq += __shfl_xor(sq, o, 32);
  }
  const float mean = s * (1.0f / HDIM);
  const float var  = sq * (1.0f / HDIM) - mean * mean;
  const float inv  = rsqrtf(var + 1e-5f);
#pragma unroll
  for (int i = 0; i < 8; ++i) {
    int j = lane + i * 32;
    float y = (x[i] - mean) * inv * g[j] + b[j];
    if (doGelu) y = gelu_exact(y);
    out[base + j] = y;
  }
}

__global__ void k_fill_i32(int* p, int v, long n) {
  long i = (long)blockIdx.x * 256 + threadIdx.x;
  if (i < n) p[i] = v;
}
__global__ void k_fill_f32(float* p, float v, long n) {
  long i = (long)blockIdx.x * 256 + threadIdx.x;
  if (i < n) p[i] = v;
}

// ---------------------------------------------------------------------------
// Edge attention
// ---------------------------------------------------------------------------
__global__ __launch_bounds__(256) void k_edge_score(
    const int* __restrict__ ei, const float* __restrict__ ea,
    const float* __restrict__ q, const float* __restrict__ kf,
    const float* __restrict__ We, const float* __restrict__ be,
    float* __restrict__ score, int* __restrict__ mxi) {
  long idx = (long)blockIdx.x * 256 + threadIdx.x;
  if (idx >= (long)E_EDGES * NHEADS) return;
  const int e  = (int)(idx >> 3);
  const int hd = (int)(idx & 7);
  const int s = ei[e];
  const int d = ei[E_EDGES + e];
  const float4* qp = (const float4*)(q  + (long)d * HDIM + hd * HEAD_D);
  const float4* kp = (const float4*)(kf + (long)s * HDIM + hd * HEAD_D);
  float acc = 0.0f;
#pragma unroll
  for (int i = 0; i < HEAD_D / 4; ++i) {
    const float4 a = qp[i], b = kp[i];
    acc += a.x * b.x + a.y * b.y + a.z * b.z + a.w * b.w;
  }
  const float sc = acc * INV_SCALE + ea[e * 2] * We[hd] + ea[e * 2 + 1] * We[8 + hd] + be[hd];
  score[idx] = sc;
  atomicMax(&mxi[d * NHEADS + hd], fenc(sc));
}

__global__ __launch_bounds__(256) void k_edge_ex(
    const int* __restrict__ ei, float* __restrict__ score,
    const int* __restrict__ mxi, float* __restrict__ denom) {
  long idx = (long)blockIdx.x * 256 + threadIdx.x;
  if (idx >= (long)E_EDGES * NHEADS) return;
  const int e  = (int)(idx >> 3);
  const int hd = (int)(idx & 7);
  const int d = ei[E_EDGES + e];
  const int m = mxi[d * NHEADS + hd];
  const float mf = (m == (int)0x80000000) ? 0.0f : fdec(m);
  const float ex = expf(score[idx] - mf);
  score[idx] = ex;
  atomicAdd(&denom[d * NHEADS + hd], ex);
}

// 4 edges per block; 64 threads/edge, each thread a float4 of the feature dim
__global__ __launch_bounds__(256) void k_edge_msg(
    const int* __restrict__ ei, const float* __restrict__ ex,
    const float* __restrict__ denom, const float* __restrict__ v,
    float* __restrict__ agg) {
  const int e   = blockIdx.x * 4 + (threadIdx.x >> 6);
  const int j0  = (threadIdx.x & 63) * 4;
  const int hd  = j0 >> 5;
  const int s = ei[e];
  const int d = ei[E_EDGES + e];
  const float w = ex[e * NHEADS + hd] / (denom[d * NHEADS + hd] + 1e-16f);
  const float4 vv = *(const float4*)(v + (long)s * HDIM + j0);
  float* ap = agg + (long)d * HDIM + j0;
  atomicAdd(ap + 0, w * vv.x);
  atomicAdd(ap + 1, w * vv.y);
  atomicAdd(ap + 2, w * vv.z);
  atomicAdd(ap + 3, w * vv.w);
}

// ---------------------------------------------------------------------------
// Classifier tail: (N,128) @ (128,2) + softmax; writes logits then probs
// ---------------------------------------------------------------------------
__global__ __launch_bounds__(256) void k_cls(
    const float* __restrict__ c1, const float* __restrict__ W2,
    const float* __restrict__ b2, float* __restrict__ outp) {
  const int n = blockIdx.x * 256 + threadIdx.x;
  if (n >= N_NODES) return;
  const float* p = c1 + (long)n * 128;
  float l0 = b2[0], l1 = b2[1];
#pragma unroll 4
  for (int i = 0; i < 128; ++i) {
    const float v = p[i];
    l0 += v * W2[i * 2];
    l1 += v * W2[i * 2 + 1];
  }
  const float m  = fmaxf(l0, l1);
  const float e0 = expf(l0 - m), e1 = expf(l1 - m);
  const float inv = 1.0f / (e0 + e1);
  outp[n * 2]     = l0;
  outp[n * 2 + 1] = l1;
  outp[2 * N_NODES + n * 2]     = e0 * inv;
  outp[2 * N_NODES + n * 2 + 1] = e1 * inv;
}

// ---------------------------------------------------------------------------
extern "C" void kernel_launch(void* const* d_in, const int* in_sizes, int n_in,
                              void* d_out, int out_size, void* d_ws, size_t ws_size,
                              hipStream_t stream) {
  const float* x     = (const float*)d_in[0];
  const int*   ei    = (const int*)d_in[1];
  const float* ea    = (const float*)d_in[2];
  const float* in_W  = (const float*)d_in[3];
  const float* in_b  = (const float*)d_in[4];
  const float* in_g  = (const float*)d_in[5];
  const float* in_bb = (const float*)d_in[6];
  const float* Wq    = (const float*)d_in[7];
  const float* Wk    = (const float*)d_in[8];
  const float* Wv    = (const float*)d_in[9];
  const float* We    = (const float*)d_in[10];
  const float* be    = (const float*)d_in[11];
  const float* Wo    = (const float*)d_in[12];
  const float* bo    = (const float*)d_in[13];
  const float* ln1_g = (const float*)d_in[14];
  const float* ln1_b = (const float*)d_in[15];
  const float* ln2_g = (const float*)d_in[16];
  const float* ln2_b = (const float*)d_in[17];
  const float* f1_W  = (const float*)d_in[18];
  const float* f1_b  = (const float*)d_in[19];
  const float* f2_W  = (const float*)d_in[20];
  const float* f2_b  = (const float*)d_in[21];
  const float* c1_W  = (const float*)d_in[22];
  const float* c1_b  = (const float*)d_in[23];
  const float* c2_W  = (const float*)d_in[24];
  const float* c2_b  = (const float*)d_in[25];

  const size_t NH = (size_t)N_NODES * HDIM;  // 12.8M floats
  float* ws   = (float*)d_ws;
  float* h    = ws;            // N x H
  float* hn   = ws + NH;       // N x H (also tmp2 / c1 scratch)
  float* q    = ws + 2 * NH;   // N x H
  float* kk   = ws + 3 * NH;   // N x H
  float* v    = ws + 4 * NH;   // N x H
  float* agg  = ws + 5 * NH;   // N x H
  float* outb = ws + 6 * NH;   // N x H
  float* y1   = q;             // N x FF reuses q..agg (exactly 4*NH)
  float* score = ws + 7 * NH;                       // E x 8
  float* denom = score + (size_t)E_EDGES * NHEADS;  // N x 8
  int*   mxi   = (int*)(denom + (size_t)N_NODES * NHEADS);

  const dim3 blk(256);
  auto gemm = [&](const float* A, const float* W, const float* bias,
                  const float* resid, float* out, int M, int K, int Nout, int act) {
    dim3 grid((M + BM - 1) / BM, Nout / BN);
    k_gemm_wmma<<<grid, blk, 0, stream>>>(A, W, bias, resid, out, M, K, Nout, act);
  };
  const int lnGrid = (N_NODES + 7) / 8;
  const long eh = (long)E_EDGES * NHEADS;
  const int ehGrid = (int)((eh + 255) / 256);
  const int nhGrid = (int)(((long)N_NODES * NHEADS + 255) / 256);
  const int aggGrid = (int)((NH + 255) / 256);

  // Input projection + LN + GELU
  gemm(x, in_W, in_b, nullptr, hn, N_NODES, FDIM_IN, HDIM, 0);
  k_ln<<<lnGrid, blk, 0, stream>>>(hn, nullptr, in_g, in_bb, h, N_NODES, 1);

  for (int l = 0; l < NLAYERS; ++l) {
    const float* Wq_l = Wq + (size_t)l * HDIM * HDIM;
    const float* Wk_l = Wk + (size_t)l * HDIM * HDIM;
    const float* Wv_l = Wv + (size_t)l * HDIM * HDIM;
    const float* We_l = We + (size_t)l * 2 * NHEADS;
    const float* be_l = be + (size_t)l * NHEADS;
    const float* Wo_l = Wo + (size_t)l * HDIM * HDIM;
    const float* bo_l = bo + (size_t)l * HDIM;
    const float* g1 = ln1_g + (size_t)l * HDIM, *b1 = ln1_b + (size_t)l * HDIM;
    const float* g2 = ln2_g + (size_t)l * HDIM, *b2 = ln2_b + (size_t)l * HDIM;
    const float* w1 = f1_W + (size_t)l * HDIM * FFDIM, *bf1 = f1_b + (size_t)l * FFDIM;
    const float* w2 = f2_W + (size_t)l * FFDIM * HDIM, *bf2 = f2_b + (size_t)l * HDIM;

    k_ln<<<lnGrid, blk, 0, stream>>>(h, nullptr, g1, b1, hn, N_NODES, 0);
    gemm(hn, Wq_l, nullptr, nullptr, q,  N_NODES, HDIM, HDIM, 0);
    gemm(hn, Wk_l, nullptr, nullptr, kk, N_NODES, HDIM, HDIM, 0);
    gemm(hn, Wv_l, nullptr, nullptr, v,  N_NODES, HDIM, HDIM, 0);

    k_fill_i32<<<nhGrid, blk, 0, stream>>>(mxi, (int)0x80000000, (long)N_NODES * NHEADS);
    k_fill_f32<<<nhGrid, blk, 0, stream>>>(denom, 0.0f, (long)N_NODES * NHEADS);
    k_fill_f32<<<aggGrid, blk, 0, stream>>>(agg, 0.0f, (long)NH);

    k_edge_score<<<ehGrid, blk, 0, stream>>>(ei, ea, q, kk, We_l, be_l, score, mxi);
    k_edge_ex<<<ehGrid, blk, 0, stream>>>(ei, score, mxi, denom);
    k_edge_msg<<<E_EDGES / 4, blk, 0, stream>>>(ei, score, denom, v, agg);

    gemm(agg, Wo_l, bo_l, nullptr, hn, N_NODES, HDIM, HDIM, 0);           // tmp2
    k_ln<<<lnGrid, blk, 0, stream>>>(hn, h, g2, b2, outb, N_NODES, 0);    // LN(tmp2+h)
    gemm(outb, w1, bf1, nullptr, y1, N_NODES, HDIM, FFDIM, 1);            // GELU
    gemm(y1, w2, bf2, outb, h, N_NODES, FFDIM, HDIM, 0);                  // h = out+ffn
  }

  // Classifier
  gemm(h, c1_W, c1_b, nullptr, hn, N_NODES, HDIM, 128, 1);  // c1 in hn
  k_cls<<<(N_NODES + 255) / 256, blk, 0, stream>>>(hn, c2_W, c2_b, (float*)d_out);
}
